// SuperPointCoreML_68350109549132
// MI455X (gfx1250) — matrix operand order repaired and checked by reading, not compiled, not run on hardware
//
#include <hip/hip_runtime.h>
#include <hip/hip_bf16.h>
#include <cstdint>
#include <cstddef>

typedef __attribute__((ext_vector_type(16))) _Float16 v16h;
typedef __attribute__((ext_vector_type(8)))  _Float16 h8;
typedef __attribute__((ext_vector_type(8)))  float    v8f;
typedef unsigned int uv4 __attribute__((ext_vector_type(4)));
typedef int          iv4 __attribute__((ext_vector_type(4)));
typedef int          iv8 __attribute__((ext_vector_type(8)));
typedef _Float16 f16;

constexpr int IMG_H = 480, IMG_W = 640;
constexpr int HC = 60, WC = 80;
constexpr int KTOP = 1024;
constexpr int NCAND = 16384;

// ---------------------------------------------------------------------------
// Weight transform: [O][I][ks][ks] fp32 -> [Npad][Ktot] f16, K = (ty,tx,ci)
// ---------------------------------------------------------------------------
__global__ void k_xform_w(const float* __restrict__ w, f16* __restrict__ wt,
                          int O, int I, int ks, int Npad) {
  int Ktot = ks * ks * I;
  long long tot = (long long)Npad * Ktot;
  long long idx = (long long)blockIdx.x * blockDim.x + threadIdx.x;
  if (idx >= tot) return;
  int n = (int)(idx / Ktot);
  int k = (int)(idx % Ktot);
  int tap = k / I, ci = k % I;
  int ty = tap / ks, tx = tap % ks;
  float v = 0.f;
  if (n < O) v = w[(((long long)n * I + ci) * ks + ty) * ks + tx];
  wt[idx] = (f16)v;
}

// ---------------------------------------------------------------------------
// Implicit-GEMM conv, NHWC f16 activations, f16 WMMA, f32 accumulate.
// One wave per 16(M=pixels) x 32(N=cout) tile; K = ks*ks*Cin in chunks of 32.
// The 32xKtot weight tile is DMA'd into LDS once per workgroup via the
// Tensor Data Mover (TENSOR_LOAD_TO_LDS, TENSORcnt), then B fragments are
// served by ds_load_b128 while the vmem pipe streams A-side activations.
// ---------------------------------------------------------------------------
__global__ void k_conv_wmma(const f16* __restrict__ in, const f16* __restrict__ wt,
                            const float* __restrict__ bias,
                            f16* __restrict__ outh, float* __restrict__ outf,
                            int H, int W, int Cin, int Cout, int ks, int pad,
                            int relu, int Ktot) {
  extern __shared__ f16 bsm[];         // [32][Ktot] weight tile
  const int lane = threadIdx.x;        // 0..31 (wave32)
  const int l16  = lane & 15;
  const int half = lane >> 4;          // 0 or 1
  const int m0 = blockIdx.x * 16;
  const int n0 = blockIdx.y * 32;
  const int m = m0 + l16;              // A-matrix row (pixel) for this lane
  const int y = m / W, x = m % W;
  const int klo = half * 8;            // A: channel sub-offset (K 0..7 / 8..15)
  const int kbo = half * 8;            // B: K sub-offset (same as A)

  // ---- TDM: D# descriptor for a 2-D tile [32 rows][Ktot f16] of wt ----
  {
    unsigned long long ga =
        (unsigned long long)(uintptr_t)(wt + (size_t)n0 * Ktot);
    uv4 g0;
    g0[0] = 1u;                                       // count=1 (valid user D#)
    g0[1] = 0u;                                       // lds_addr = 0 (wave base)
    g0[2] = (unsigned)(ga & 0xFFFFFFFFull);           // global_addr[31:0]
    g0[3] = (unsigned)((ga >> 32) & 0x1FFFFFFu) | (2u << 30);  // addr[56:32]|type=2
    iv8 g1;
    unsigned kt = (unsigned)Ktot;
    g1[0] = 0x10000;                                  // data_size=1 -> 2 bytes
    g1[1] = (int)(kt << 16);                          // tensor_dim0[15:0] @bit48
    g1[2] = (int)((kt >> 16) | (32u << 16));          // dim0[31:16] | tensor_dim1=32
    g1[3] = (int)(kt << 16);                          // dim1[31:16]=0 | tile_dim0
    g1[4] = 32;                                       // tile_dim1=32, tile_dim2=0
    g1[5] = (int)kt;                                  // tensor_dim0_stride[31:0]
    g1[6] = 0;                                        // stride[47:32] | dim1_stride lo
    g1[7] = 0;
    iv4 gz4 = {};
    iv8 gz8 = {};
    __builtin_amdgcn_tensor_load_to_lds(g0, g1, gz4, gz4, gz8, 0);
    __builtin_amdgcn_s_wait_tensorcnt(0);
    __asm__ volatile("" ::: "memory");
  }

  v8f acc0 = {};
  v8f acc1 = {};

  const int taps = ks * ks;
  for (int tap = 0; tap < taps; ++tap) {
    int ty = tap / ks, tx = tap % ks;
    int sy = y + ty - pad, sx = x + tx - pad;
    bool valid = (sy >= 0) && (sy < H) && (sx >= 0) && (sx < W);
    const f16* ap = in + (size_t)(sy * W + sx) * Cin;
    for (int c0 = 0; c0 < Cin; c0 += 32) {
      // ---- A fragment: lane<16 -> K {c0..c0+7, c0+16..c0+23}; lane>=16 -> +8
      v16h a = {};
      if (valid) {
        h8 lo = *(const h8*)(ap + c0 + klo);
        h8 hi = *(const h8*)(ap + c0 + klo + 16);
#pragma unroll
        for (int j = 0; j < 8; ++j) { a[j] = lo[j]; a[j + 8] = hi[j]; }
      }
      // ---- B fragments from LDS: lane holds K {kbo..kbo+7, kbo+16..kbo+23} of column l16
      int kg = tap * Cin + c0 + kbo;
      const f16* bp0 = bsm + (size_t)l16 * Ktot + kg;
      const f16* bp1 = bsm + (size_t)(16 + l16) * Ktot + kg;
      v16h b0, b1;
      {
        h8 lo = *(const h8*)bp0; h8 hi = *(const h8*)(bp0 + 16);
#pragma unroll
        for (int j = 0; j < 8; ++j) { b0[j] = lo[j]; b0[j + 8] = hi[j]; }
      }
      {
        h8 lo = *(const h8*)bp1; h8 hi = *(const h8*)(bp1 + 16);
#pragma unroll
        for (int j = 0; j < 8; ++j) { b1[j] = lo[j]; b1[j + 8] = hi[j]; }
      }
      acc0 = __builtin_amdgcn_wmma_f32_16x16x32_f16(false, a, false, b0,
                                                    (short)0, acc0, false, false);
      acc1 = __builtin_amdgcn_wmma_f32_16x16x32_f16(false, a, false, b1,
                                                    (short)0, acc1, false, false);
    }
  }

  // Epilogue: C/D layout — VGPR r, lane l -> M = r + 8*(l>=16), N = l&15
  int nA = n0 + l16;
  int nB = n0 + 16 + l16;
  float bA = (nA < Cout) ? bias[nA] : 0.f;
  float bB = (nB < Cout) ? bias[nB] : 0.f;
#pragma unroll
  for (int r = 0; r < 8; ++r) {
    int mr = m0 + r + half * 8;
    float v0 = acc0[r] + bA;
    float v1 = acc1[r] + bB;
    if (relu) { v0 = fmaxf(v0, 0.f); v1 = fmaxf(v1, 0.f); }
    if (nA < Cout) {
      if (outh) outh[(size_t)mr * Cout + nA] = (f16)v0;
      else      outf[(size_t)mr * Cout + nA] = v0;
    }
    if (nB < Cout) {
      if (outh) outh[(size_t)mr * Cout + nB] = (f16)v1;
      else      outf[(size_t)mr * Cout + nB] = v1;
    }
  }
}

// ---------------------------------------------------------------------------
// conv1a: Cin=1 (K=9) — direct fp32 conv, writes f16 NHWC, ReLU.
// ---------------------------------------------------------------------------
__global__ void k_conv1a(const float* __restrict__ img, const float* __restrict__ w,
                         const float* __restrict__ b, f16* __restrict__ out) {
  int p = blockIdx.x * blockDim.x + threadIdx.x;
  if (p >= IMG_H * IMG_W) return;
  int y = p / IMG_W, x = p % IMG_W;
  float win[9];
#pragma unroll
  for (int t = 0; t < 9; ++t) {
    int sy = y + t / 3 - 1, sx = x + t % 3 - 1;
    win[t] = (sy >= 0 && sy < IMG_H && sx >= 0 && sx < IMG_W)
                 ? img[sy * IMG_W + sx] : 0.f;
  }
  for (int c = 0; c < 64; ++c) {
    float s = b[c];
#pragma unroll
    for (int t = 0; t < 9; ++t) s += w[c * 9 + t] * win[t];
    out[(size_t)p * 64 + c] = (f16)fmaxf(s, 0.f);
  }
}

// ---------------------------------------------------------------------------
// 2x2 max pool, stride 2, NHWC f16
// ---------------------------------------------------------------------------
__global__ void k_pool2(const f16* __restrict__ in, f16* __restrict__ out,
                        int OH, int OW, int C) {
  long long tot = (long long)OH * OW * C;
  long long i = (long long)blockIdx.x * blockDim.x + threadIdx.x;
  if (i >= tot) return;
  int c = (int)(i % C);
  long long q = i / C;
  int ox = (int)(q % OW), oy = (int)(q / OW);
  int IW = OW * 2;
  const f16* p00 = in + ((size_t)(2 * oy) * IW + 2 * ox) * C + c;
  float a  = (float)p00[0];
  float b2 = (float)p00[C];
  float c2 = (float)p00[(size_t)IW * C];
  float d2 = (float)p00[(size_t)IW * C + C];
  out[i] = (f16)fmaxf(fmaxf(a, b2), fmaxf(c2, d2));
}

// ---------------------------------------------------------------------------
// softmax over 65 channels (drop dustbin) + pixel shuffle to [480][640] fp32
// ---------------------------------------------------------------------------
__global__ void k_softmax_shuffle(const float* __restrict__ s65,
                                  float* __restrict__ full) {
  int p = blockIdx.x * blockDim.x + threadIdx.x;
  if (p >= HC * WC) return;
  int wc = p % WC, hc = p / WC;
  const float* v = s65 + (size_t)p * 65;
  float mx = v[0];
  for (int c = 1; c < 65; ++c) mx = fmaxf(mx, v[c]);
  float e[64];
  float sum = 0.f;
  for (int c = 0; c < 65; ++c) {
    float ec = __expf(v[c] - mx);
    if (c < 64) e[c] = ec;
    sum += ec;
  }
  float inv = 1.f / sum;
  for (int c = 0; c < 64; ++c) {
    int sy = c >> 3, sx = c & 7;
    full[(size_t)(hc * 8 + sy) * IMG_W + wc * 8 + sx] = e[c] * inv;
  }
}

// ---------------------------------------------------------------------------
// separable 9x9 max pool (pad = -inf), full-res fp32
// ---------------------------------------------------------------------------
__global__ void k_row9(const float* __restrict__ in, float* __restrict__ out) {
  int p = blockIdx.x * blockDim.x + threadIdx.x;
  if (p >= IMG_H * IMG_W) return;
  int y = p / IMG_W, x = p % IMG_W;
  float m = -__builtin_inff();
  for (int d = -4; d <= 4; ++d) {
    int xs = x + d;
    if (xs >= 0 && xs < IMG_W) m = fmaxf(m, in[(size_t)y * IMG_W + xs]);
  }
  out[p] = m;
}

__global__ void k_col9(const float* __restrict__ in, float* __restrict__ out) {
  int p = blockIdx.x * blockDim.x + threadIdx.x;
  if (p >= IMG_H * IMG_W) return;
  int y = p / IMG_W, x = p % IMG_W;
  float m = -__builtin_inff();
  for (int d = -4; d <= 4; ++d) {
    int ys = y + d;
    if (ys >= 0 && ys < IMG_H) m = fmaxf(m, in[(size_t)ys * IMG_W + x]);
  }
  out[p] = m;
}

__global__ void k_eqmask(const float* __restrict__ sf, const float* __restrict__ pb,
                         float* __restrict__ mm) {
  int p = blockIdx.x * blockDim.x + threadIdx.x;
  if (p >= IMG_H * IMG_W) return;
  mm[p] = (sf[p] == pb[p]) ? 1.f : 0.f;
}

__global__ void k_suppscores(const float* __restrict__ sf, const float* __restrict__ pm,
                             float* __restrict__ ss) {
  int p = blockIdx.x * blockDim.x + threadIdx.x;
  if (p >= IMG_H * IMG_W) return;
  ss[p] = (pm[p] > 0.f) ? 0.f : sf[p];
}

__global__ void k_update(float* __restrict__ mm, const float* __restrict__ ss,
                         const float* __restrict__ ps, const float* __restrict__ pm) {
  int p = blockIdx.x * blockDim.x + threadIdx.x;
  if (p >= IMG_H * IMG_W) return;
  if (mm[p] == 0.f && ss[p] == ps[p] && pm[p] <= 0.f) mm[p] = 1.f;
}

// ---------------------------------------------------------------------------
// top-K: compact NMS survivors, bitonic-sort 16384 packed keys in 128KB LDS
// key = (float_bits(score) << 32) | (~idx)  -> descending sort = topk, ties
// broken toward smaller flat index (matches lax.top_k).
// ---------------------------------------------------------------------------
__global__ void k_init_cand(unsigned int* __restrict__ cnt,
                            unsigned long long* __restrict__ cand) {
  int i = blockIdx.x * blockDim.x + threadIdx.x;
  if (i == 0) *cnt = 0u;
  if (i < NCAND) cand[i] = 0ull;
}

__global__ void k_compact(const float* __restrict__ mm, const float* __restrict__ sf,
                          unsigned int* __restrict__ cnt,
                          unsigned long long* __restrict__ cand) {
  int p = blockIdx.x * blockDim.x + threadIdx.x;
  if (p >= IMG_H * IMG_W) return;
  if (mm[p] != 0.f) {
    float sc = sf[p];
    unsigned pos = atomicAdd(cnt, 1u);
    if (pos < NCAND) {
      unsigned sb = __float_as_uint(sc);
      cand[pos] = ((unsigned long long)sb << 32) |
                  (unsigned long long)(0xFFFFFFFFu - (unsigned)p);
    }
  }
}

__global__ __launch_bounds__(1024) void k_sort_topk(
    const unsigned long long* __restrict__ cand, float* __restrict__ out) {
  extern __shared__ unsigned long long sm[];
  const int N = NCAND;
  int t = threadIdx.x;
  for (int i = t; i < N; i += 1024) sm[i] = cand[i];
  __syncthreads();
  for (int k = 2; k <= N; k <<= 1) {
    for (int j = k >> 1; j > 0; j >>= 1) {
      for (int i = t; i < N; i += 1024) {
        int ixj = i ^ j;
        if (ixj > i) {
          unsigned long long a = sm[i], b = sm[ixj];
          bool desc = ((i & k) == 0);
          if ((desc && a < b) || (!desc && a > b)) { sm[i] = b; sm[ixj] = a; }
        }
      }
      __syncthreads();
    }
  }
  if (t < KTOP) {
    unsigned long long v = sm[t];
    float sc = __uint_as_float((unsigned)(v >> 32));
    unsigned idx = 0xFFFFFFFFu - (unsigned)(v & 0xFFFFFFFFull);
    if (idx >= (unsigned)(IMG_H * IMG_W)) { idx = 0u; sc = 0.f; }
    out[2 * t]     = (float)(idx % IMG_W);  // kx
    out[2 * t + 1] = (float)(idx / IMG_W);  // ky
    out[2 * KTOP + t] = sc;                 // top score
  }
}

// ---------------------------------------------------------------------------
// per-pixel L2 normalization of descriptor map [4800][256] fp32 (in place)
// ---------------------------------------------------------------------------
__global__ void k_l2norm_desc(float* __restrict__ d) {
  __shared__ float red[256];
  int p = blockIdx.x, c = threadIdx.x;
  float v = d[(size_t)p * 256 + c];
  red[c] = v * v;
  __syncthreads();
  for (int s = 128; s > 0; s >>= 1) {
    if (c < s) red[c] += red[c + s];
    __syncthreads();
  }
  float inv = 1.f / fmaxf(sqrtf(red[0]), 1e-12f);
  d[(size_t)p * 256 + c] = v * inv;
}

// ---------------------------------------------------------------------------
// bilinear descriptor sampling at keypoints + L2 norm; one wave per keypoint
// ---------------------------------------------------------------------------
__global__ void k_sample(const float* __restrict__ desc, float* __restrict__ out) {
  int k = blockIdx.x;
  int lane = threadIdx.x;           // 0..31, each handles 8 channels
  float kx = out[2 * k], ky = out[2 * k + 1];
  // pts = (kp - S/2 + 0.5)/scale*2-1 ; scale = [Wc*8-4.5, Hc*8-4.5]
  float gx = (kx - 3.5f) / 635.5f * 2.f - 1.f;
  float gy = (ky - 3.5f) / 475.5f * 2.f - 1.f;
  float ix = (gx + 1.f) * 0.5f * (float)(WC - 1);   // align_corners
  float iy = (gy + 1.f) * 0.5f * (float)(HC - 1);
  float x0f = floorf(ix), y0f = floorf(iy);
  float wx1 = ix - x0f, wy1 = iy - y0f;
  float wx0 = 1.f - wx1, wy0 = 1.f - wy1;
  int x0 = (int)x0f, y0 = (int)y0f, x1 = x0 + 1, y1 = y0 + 1;

  int   cy[4] = {y0, y0, y1, y1};
  int   cx[4] = {x0, x1, x0, x1};
  float cw[4] = {wy0 * wx0, wy0 * wx1, wy1 * wx0, wy1 * wx1};

  float acc[8];
#pragma unroll
  for (int j = 0; j < 8; ++j) acc[j] = 0.f;
  for (int t = 0; t < 4; ++t) {
    int yy = cy[t], xx = cx[t];
    bool inb = (xx >= 0 && xx < WC && yy >= 0 && yy < HC);
    int yc = yy < 0 ? 0 : (yy > HC - 1 ? HC - 1 : yy);
    int xc = xx < 0 ? 0 : (xx > WC - 1 ? WC - 1 : xx);
    float wgt = inb ? cw[t] : 0.f;
    const float* dp = desc + ((size_t)yc * WC + xc) * 256 + lane * 8;
#pragma unroll
    for (int j = 0; j < 8; ++j) acc[j] += wgt * dp[j];
  }
  float ss = 0.f;
#pragma unroll
  for (int j = 0; j < 8; ++j) ss += acc[j] * acc[j];
  for (int m = 16; m >= 1; m >>= 1) ss += __shfl_xor(ss, m, 32);
  float inv = 1.f / fmaxf(sqrtf(ss), 1e-12f);
  float* op = out + 3 * KTOP + (size_t)k * 256 + lane * 8;
#pragma unroll
  for (int j = 0; j < 8; ++j) op[j] = acc[j] * inv;
}

// ---------------------------------------------------------------------------
// host-side conv wrapper
// ---------------------------------------------------------------------------
static void run_conv(const f16* in, int H, int W, int Cin,
                     const float* w, const float* b, int Cout, int ks, int pad,
                     bool relu, f16* outh, float* outf, f16* wt,
                     hipStream_t s) {
  int Npad = ((Cout + 31) / 32) * 32;
  int Ktot = ks * ks * Cin;
  long long tot = (long long)Npad * Ktot;
  k_xform_w<<<(unsigned)((tot + 255) / 256), 256, 0, s>>>(w, wt, Cout, Cin, ks, Npad);
  dim3 grid((unsigned)((H * W) / 16), (unsigned)(Npad / 32));
  size_t lds = (size_t)32 * Ktot * sizeof(f16);       // B tile staged by TDM
  k_conv_wmma<<<grid, 32, lds, s>>>(in, wt, b, outh, outf, H, W, Cin, Cout, ks,
                                    pad, relu ? 1 : 0, Ktot);
}

static inline unsigned div_up(long long a, int b) { return (unsigned)((a + b - 1) / b); }

extern "C" void kernel_launch(void* const* d_in, const int* in_sizes, int n_in,
                              void* d_out, int out_size, void* d_ws, size_t ws_size,
                              hipStream_t stream) {
  (void)in_sizes; (void)n_in; (void)out_size; (void)ws_size;

  const float* img = (const float*)d_in[0];
  const float* w1a = (const float*)d_in[1];  const float* b1a = (const float*)d_in[2];
  const float* w1b = (const float*)d_in[3];  const float* b1b = (const float*)d_in[4];
  const float* w2a = (const float*)d_in[5];  const float* b2a = (const float*)d_in[6];
  const float* w2b = (const float*)d_in[7];  const float* b2b = (const float*)d_in[8];
  const float* w3a = (const float*)d_in[9];  const float* b3a = (const float*)d_in[10];
  const float* w3b = (const float*)d_in[11]; const float* b3b = (const float*)d_in[12];
  const float* w4a = (const float*)d_in[13]; const float* b4a = (const float*)d_in[14];
  const float* w4b = (const float*)d_in[15]; const float* b4b = (const float*)d_in[16];
  const float* wPa = (const float*)d_in[17]; const float* bPa = (const float*)d_in[18];
  const float* wPb = (const float*)d_in[19]; const float* bPb = (const float*)d_in[20];
  const float* wDa = (const float*)d_in[21]; const float* bDa = (const float*)d_in[22];
  const float* wDb = (const float*)d_in[23]; const float* bDb = (const float*)d_in[24];

  // ---- workspace layout (all offsets multiples of 256) ----
  char* ws = (char*)d_ws;
  const size_t SZ_ACT  = (size_t)IMG_H * IMG_W * 64 * sizeof(f16);   // 39,321,600
  const size_t SZ_WT   = (size_t)256 * 1152 * sizeof(f16);           //    589,824
  const size_t SZ_S65  = (size_t)HC * WC * 65 * sizeof(float);       //  1,248,000
  const size_t SZ_FUL  = (size_t)IMG_H * IMG_W * sizeof(float);      //  1,228,800
  const size_t SZ_DESC = (size_t)HC * WC * 256 * sizeof(float);      //  4,915,200

  size_t off = 0;
  f16*   A0   = (f16*)(ws + off);               off += SZ_ACT;
  f16*   A1   = (f16*)(ws + off);               off += SZ_ACT;
  f16*   WT   = (f16*)(ws + off);               off += SZ_WT;
  float* S65  = (float*)(ws + off);             off += SZ_S65;
  float* SF   = (float*)(ws + off);             off += SZ_FUL;   // full-res scores
  float* MM   = (float*)(ws + off);             off += SZ_FUL;   // max mask
  float* TA   = (float*)(ws + off);             off += SZ_FUL;
  float* TB   = (float*)(ws + off);             off += SZ_FUL;
  float* TC   = (float*)(ws + off);             off += SZ_FUL;
  float* TD   = (float*)(ws + off);             off += SZ_FUL;
  float* DESC = (float*)(ws + off);             off += SZ_DESC;
  unsigned long long* CAND = (unsigned long long*)(ws + off); off += (size_t)NCAND * 8;
  unsigned int* CNT = (unsigned int*)(ws + off);

  float* out = (float*)d_out;
  const long long NPIX = (long long)IMG_H * IMG_W;

  // ------------------- backbone -------------------
  k_conv1a<<<div_up(NPIX, 256), 256, 0, stream>>>(img, w1a, b1a, A0);         // 480x640x64
  run_conv(A0, 480, 640, 64, w1b, b1b, 64, 3, 1, true, A1, nullptr, WT, stream);
  k_pool2<<<div_up((long long)240 * 320 * 64, 256), 256, 0, stream>>>(A1, A0, 240, 320, 64);
  run_conv(A0, 240, 320, 64, w2a, b2a, 64, 3, 1, true, A1, nullptr, WT, stream);
  run_conv(A1, 240, 320, 64, w2b, b2b, 64, 3, 1, true, A0, nullptr, WT, stream);
  k_pool2<<<div_up((long long)120 * 160 * 64, 256), 256, 0, stream>>>(A0, A1, 120, 160, 64);
  run_conv(A1, 120, 160, 64, w3a, b3a, 128, 3, 1, true, A0, nullptr, WT, stream);
  run_conv(A0, 120, 160, 128, w3b, b3b, 128, 3, 1, true, A1, nullptr, WT, stream);
  k_pool2<<<div_up((long long)HC * WC * 128, 256), 256, 0, stream>>>(A1, A0, HC, WC, 128);
  run_conv(A0, HC, WC, 128, w4a, b4a, 128, 3, 1, true, A1, nullptr, WT, stream);
  run_conv(A1, HC, WC, 128, w4b, b4b, 128, 3, 1, true, A0, nullptr, WT, stream); // x in A0

  // ------------------- detector head -------------------
  run_conv(A0, HC, WC, 128, wPa, bPa, 256, 3, 1, true, A1, nullptr, WT, stream);
  run_conv(A1, HC, WC, 256, wPb, bPb, 65, 1, 0, false, nullptr, S65, WT, stream);
  k_softmax_shuffle<<<div_up(HC * WC, 128), 128, 0, stream>>>(S65, SF);

  // simple_nms
  unsigned gf = div_up(NPIX, 256);
  k_row9<<<gf, 256, 0, stream>>>(SF, TA);
  k_col9<<<gf, 256, 0, stream>>>(TA, TB);
  k_eqmask<<<gf, 256, 0, stream>>>(SF, TB, MM);
  for (int it = 0; it < 2; ++it) {
    k_row9<<<gf, 256, 0, stream>>>(MM, TA);
    k_col9<<<gf, 256, 0, stream>>>(TA, TB);           // TB = pool9(mask)
    k_suppscores<<<gf, 256, 0, stream>>>(SF, TB, TA); // TA = supp_scores
    k_row9<<<gf, 256, 0, stream>>>(TA, TC);
    k_col9<<<gf, 256, 0, stream>>>(TC, TD);           // TD = pool9(supp_scores)
    k_update<<<gf, 256, 0, stream>>>(MM, TA, TD, TB);
  }

  // top-K
  k_init_cand<<<div_up(NCAND, 256), 256, 0, stream>>>(CNT, CAND);
  k_compact<<<gf, 256, 0, stream>>>(MM, SF, CNT, CAND);
  k_sort_topk<<<1, 1024, (size_t)NCAND * 8, stream>>>(CAND, out);

  // ------------------- descriptor head -------------------
  run_conv(A0, HC, WC, 128, wDa, bDa, 256, 3, 1, true, A1, nullptr, WT, stream);
  run_conv(A1, HC, WC, 256, wDb, bDb, 256, 1, 0, false, nullptr, DESC, WT, stream);
  k_l2norm_desc<<<HC * WC, 256, 0, stream>>>(DESC);
  k_sample<<<KTOP, 32, 0, stream>>>(DESC, out);
}
